// SnapFormerHead_63969242907165
// MI455X (gfx1250) — matrix operand order repaired and checked by reading, not compiled, not run on hardware
//
#include <hip/hip_runtime.h>
#include <hip/hip_bf16.h>
#include <math.h>

// ---------------------------------------------------------------------------
// Problem constants (from reference)
// ---------------------------------------------------------------------------
#define BB   8
#define TT   2048
#define DD   1024
#define HH   512
#define H2   256
#define KK   24

#define TP   (TT + 2)          // time-padded rows for conv halo
#define NT_TILES (TT / 16)     // 128 tiles of 16 time steps

typedef _Float16 half16 __attribute__((ext_vector_type(16)));
typedef _Float16 half8  __attribute__((ext_vector_type(8)));
typedef float    f32x8  __attribute__((ext_vector_type(8)));

// Workspace layout (bytes)
#define FEH_BYTES    ((size_t)BB * TP * DD * 2)            // f16 [B, T+2, D]
#define XT_BYTES     ((size_t)BB * TT * HH * 2)            // f16 [B, T, H] (post-SiLU)
#define WPACK_BYTES  ((size_t)3 * (HH/16) * (DD/32) * 512 * 2)   // conv A-fragments
#define W1PACK_BYTES ((size_t)(KK*H2/16) * (HH/32) * 512 * 2)    // head A-fragments

__device__ __forceinline__ float silu_f(float v) {
    return v * (1.0f / (1.0f + __expf(-v)));
}
__device__ __forceinline__ float sigmoid_f(float v) {
    return 1.0f / (1.0f + __expf(-v));
}

// CDNA5 async global->LDS copy (VGLOBAL encoding, tracked by ASYNCcnt).
// ldsoff: byte offset in LDS (VDST), gaddr: 64-bit global address (VADDR).
__device__ __forceinline__ void async_copy_b128(unsigned ldsoff,
                                                const void* gaddr) {
    unsigned long long ga = (unsigned long long)(uintptr_t)gaddr;
    asm volatile("global_load_async_to_lds_b128 %0, %1, off"
                 :: "v"(ldsoff), "v"(ga)
                 : "memory");
}
__device__ __forceinline__ void async_wait0() {
    asm volatile("s_wait_asynccnt 0x0" ::: "memory");
}
// Low 32 bits of a generic pointer to LDS == LDS byte offset.
__device__ __forceinline__ unsigned lds_offset_of(const void* p) {
    return (unsigned)(uintptr_t)p;
}

// ---------------------------------------------------------------------------
// Kernel 0: cast frame embeddings f32 [B,T,D] -> f16 [B, T+2, D], zero halo
// ---------------------------------------------------------------------------
__global__ void cast_fe_kernel(const float* __restrict__ fe,
                               _Float16* __restrict__ feh) {
    size_t i = (size_t)blockIdx.x * blockDim.x + threadIdx.x;
    size_t total = (size_t)BB * TP * DD;
    if (i >= total) return;
    size_t d  = i % DD;
    size_t tp = (i / DD) % TP;
    size_t b  = i / ((size_t)DD * TP);
    _Float16 v = (_Float16)0.0f;
    if (tp >= 1 && tp <= TT)
        v = (_Float16)fe[((size_t)b * TT + (tp - 1)) * DD + d];
    feh[i] = v;
}

// ---------------------------------------------------------------------------
// Kernel 1: pack conv_w [H,D,3] into WMMA A-fragment order (f16).
// Tile order: [j][di][hi]  (K-step-major) so the GEMM's A address is LINEAR
// in the flattened K index m = j*32+di:  addr = (m*32 + hi)*512 + lane*16.
// 16-bit A 16x32 layout: lane = M(0..15) + 16*half; lane elems e=0..15 map to
// K = 16*half + e (contiguous order packed here; B uses the same).
// ---------------------------------------------------------------------------
__global__ void pack_convw_kernel(const float* __restrict__ conv_w,
                                  _Float16* __restrict__ wpack) {
    size_t i = (size_t)blockIdx.x * blockDim.x + threadIdx.x;
    size_t total = (size_t)3 * (HH/16) * (DD/32) * 512;
    if (i >= total) return;
    int e    = (int)(i & 15);
    int lane = (int)((i >> 4) & 31);
    size_t tile = i >> 9;
    int hi = (int)(tile % (HH/16));
    int di = (int)((tile / (HH/16)) % (DD/32));
    int j  = (int)(tile / ((size_t)(HH/16) * (DD/32)));
    int half = lane >> 4, lrow = lane & 15;
    int h = hi * 16 + lrow;
    int kl = 16 * half + e;
    int d = di * 32 + kl;
    wpack[i] = (_Float16)conv_w[((size_t)h * DD + d) * 3 + j];
}

// ---------------------------------------------------------------------------
// Kernel 2: pack W1 [K,H2,H] (viewed as [K*H2, H]) into A-fragment order.
// Tile order: [ri (K*H2/16)][ci (H/32)]  (ci inner -> linear in the K loop).
// ---------------------------------------------------------------------------
__global__ void pack_w1_kernel(const float* __restrict__ W1,
                               _Float16* __restrict__ w1pack) {
    size_t i = (size_t)blockIdx.x * blockDim.x + threadIdx.x;
    size_t total = (size_t)(KK*H2/16) * (HH/32) * 512;
    if (i >= total) return;
    int e    = (int)(i & 15);
    int lane = (int)((i >> 4) & 31);
    size_t tile = i >> 9;
    int ci = (int)(tile % (HH/32));
    int ri = (int)(tile / (HH/32));
    int half = lane >> 4, lrow = lane & 15;
    int row = ri * 16 + lrow;           // row = k*H2 + o
    int kl = 16 * half + e;
    int c = ci * 32 + kl;
    w1pack[i] = (_Float16)W1[(size_t)row * HH + c];
}

// ---------------------------------------------------------------------------
// Kernel 3 (stage A): temporal conv as 3 accumulated GEMM taps + SiLU.
// Grid: (ttile=128, b=8), block 256 (8 waves). Wave w owns h-tiles 4w..4w+3.
// fe tile (18 rows x 1024 f16) async-staged into LDS once per block.
// The K loop steps by 2; each body issues ALL its fragment loads first
// (distinct local arrays -> distinct registers), then the 8 WMMAs, so
// loadcnt drains progressively with zero loop-carried fragment registers.
// ---------------------------------------------------------------------------
#define FES_STRIDE (DD + 8)    // f16; row stride 2064B = 516 dwords -> no conflicts

__global__ void __launch_bounds__(256)
conv_silu_kernel(const _Float16* __restrict__ feh,     // [B, T+2, D]
                 const _Float16* __restrict__ wpack,   // [3][D/32][H/16][512]
                 const float*    __restrict__ conv_b,  // [H]
                 _Float16*       __restrict__ Xt) {    // [B, T, H]
    __shared__ _Float16 fs[18 * FES_STRIDE];

    const int ttile = blockIdx.x;
    const int b     = blockIdx.y;
    const int tid   = threadIdx.x;
    const int t0    = ttile * 16;

    // Async-stage 18 padded rows (t0 .. t0+17) x 1024 f16 into LDS.
    {
        const _Float16* src = feh + ((size_t)b * TP + t0) * DD;
        const unsigned fs_base = lds_offset_of(&fs[0]);
        // 18*1024 f16 = 36864 B = 2304 x b128; 256 threads x 9
#pragma unroll
        for (int it = 0; it < 9; ++it) {
            int idx = tid + it * 256;
            int row = idx >> 7;              // 128 b128 per 2048B row
            int col = (idx & 127) * 8;       // f16 column
            unsigned dst = fs_base + (unsigned)(row * FES_STRIDE + col) * 2u;
            async_copy_b128(dst, src + (size_t)row * DD + col);
        }
    }
    async_wait0();
    __syncthreads();

    const int wave = tid >> 5;
    const int lane = tid & 31;
    const int lrow = lane & 15;
    const int half = lane >> 4;

    f32x8 acc[4];
#pragma unroll
    for (int i = 0; i < 4; ++i) {
        int h0 = (wave * 4 + i) * 16;
        acc[i] = *(const f32x8*)(conv_b + h0 + 8 * half);   // bias, per-row
    }

    // Flattened K-loop: m = j*32 + ds, 96 steps, batch of 2 per body.
    const _Float16* fsl = fs + (size_t)lrow * FES_STRIDE + half * 16;
    // A address is linear in m: (m*32 + wave*4 + i) tiles of 512 f16.
    const _Float16* wpl = wpack + (size_t)(wave * 4) * 512 + lane * 16;

#pragma unroll 1
    for (int mb = 0; mb < 96; mb += 2) {
        // ---- issue ALL loads for this batch first ----
        half16 bf[2];
        half16 af[2][4];
#pragma unroll
        for (int u = 0; u < 2; ++u) {
            const int m = mb + u;
            const int j = m >> 5, ds = m & 31;
            bf[u] = *(const half16*)(fsl + (size_t)j * FES_STRIDE + ds * 32);
#pragma unroll
            for (int i = 0; i < 4; ++i)
                af[u][i] = *(const half16*)(wpl + ((size_t)m * 32 + i) * 512);
        }
        // ---- then the 8 WMMAs (progressive loadcnt drain) ----
#pragma unroll
        for (int u = 0; u < 2; ++u)
#pragma unroll
            for (int i = 0; i < 4; ++i)
                acc[i] = __builtin_amdgcn_wmma_f32_16x16x32_f16(
                    false, af[u][i], false, bf[u], (short)0, acc[i],
                    false, false);
    }

    // Epilogue: SiLU, cvt f16, one contiguous 16B store per lane per tile.
#pragma unroll
    for (int i = 0; i < 4; ++i) {
        int h0 = (wave * 4 + i) * 16;
        half8 o;
#pragma unroll
        for (int r = 0; r < 8; ++r)
            o[r] = (_Float16)silu_f(acc[i][r]);
        size_t dst = ((size_t)b * TT + t0 + lrow) * HH + h0 + 8 * half;
        *(half8*)(Xt + dst) = o;
    }
}

// ---------------------------------------------------------------------------
// Kernel 4 (stage B): per-head GEMM [K*H2,H] x [H,16] with fused SiLU and
// W2 reduction. Grid: (ttile=128, b=8), block 256 (8 waves).
// X tile staged via async copies; K loop in batches of 4 with loads-first
// ordering inside each batch.
// ---------------------------------------------------------------------------
#define XS_STRIDE (HH + 8)   // f16; row stride 1040B = 260 dwords -> no conflicts

__global__ void __launch_bounds__(256)
heads_kernel(const _Float16* __restrict__ Xt,      // [B, T, H]
             const _Float16* __restrict__ w1pack,  // [K*H2/16][H/32][512]
             const float*    __restrict__ b1,      // [K, H2]
             const float*    __restrict__ W2,      // [K, H2]
             const float*    __restrict__ b2,      // [K]
             float*          __restrict__ frame_logits,  // [B, T, K]
             float*          __restrict__ frame_probs) { // [B, T, K]
    __shared__ _Float16 xs[16 * XS_STRIDE];

    const int ttile = blockIdx.x;
    const int b     = blockIdx.y;
    const int tid   = threadIdx.x;
    const int t0    = ttile * 16;

    // Async-stage X tile (16 t-rows x 512 f16) into LDS.
    {
        const _Float16* src = Xt + ((size_t)b * TT + t0) * HH;
        const unsigned xs_base = lds_offset_of(&xs[0]);
        // 16*512 f16 = 16384 B = 1024 x b128; 256 threads x 4
#pragma unroll
        for (int it = 0; it < 4; ++it) {
            int idx = tid + it * 256;
            int row = idx >> 6;              // 64 b128 per 1024B row
            int col = (idx & 63) * 8;        // f16 column
            unsigned dst = xs_base + (unsigned)(row * XS_STRIDE + col) * 2u;
            async_copy_b128(dst, src + (size_t)row * HH + col);
        }
    }
    async_wait0();
    __syncthreads();

    const int wave = tid >> 5;
    const int lane = tid & 31;
    const int lrow = lane & 15;
    const int half = lane >> 4;

    const _Float16* xsl = xs + (size_t)lrow * XS_STRIDE + half * 16;

    for (int k = wave; k < KK; k += 8) {
        float logit = 0.0f;   // partial for t = t0+lrow over this lane's o's
#pragma unroll 1
        for (int ot = 0; ot < (H2/16); ++ot) {
            int r0 = k * (H2/16) + ot;           // row-tile index in [0,384)
            f32x8 acc = *(const f32x8*)(b1 + (size_t)k * H2 + ot * 16 + 8 * half);
            const _Float16* apack =
                w1pack + (size_t)r0 * (HH/32) * 512 + lane * 16;
            if (ot + 1 < (H2/16))
                __builtin_prefetch(apack + (size_t)(HH/32) * 512, 0, 0);

            // 16 K-steps in 4 batches of 4: loads first, then WMMAs.
#pragma unroll 1
            for (int csb = 0; csb < (HH/32); csb += 4) {
                half16 af[4];
                half16 bf[4];
#pragma unroll
                for (int u = 0; u < 4; ++u) {
                    af[u] = *(const half16*)(apack + (size_t)(csb + u) * 512);
                    bf[u] = *(const half16*)(xsl + (csb + u) * 32);
                }
#pragma unroll
                for (int u = 0; u < 4; ++u)
                    acc = __builtin_amdgcn_wmma_f32_16x16x32_f16(
                        false, af[u], false, bf[u], (short)0, acc,
                        false, false);
            }
            // Fused SiLU + W2 dot over the 8 o-rows this lane holds.
            const float* w2p = W2 + (size_t)k * H2 + ot * 16 + 8 * half;
#pragma unroll
            for (int r = 0; r < 8; ++r)
                logit += silu_f(acc[r]) * w2p[r];
        }
        // Combine the two K-halves (lane ^ 16 holds same t, other o-range).
        logit += __shfl_xor(logit, 16, 32);
        if (half == 0) {
            float L = logit + b2[k];
            size_t o = ((size_t)b * TT + t0 + lrow) * KK + k;
            frame_logits[o] = L;
            frame_probs[o]  = sigmoid_f(L);
        }
    }
}

// ---------------------------------------------------------------------------
// Kernel 5 (stage C): clip logits = max over t, + sigmoid. Grid (K, B).
// ---------------------------------------------------------------------------
__global__ void __launch_bounds__(256)
clip_kernel(const float* __restrict__ frame_logits,  // [B, T, K]
            float* __restrict__ clip_logits,         // [B, K]
            float* __restrict__ clip_probs) {        // [B, K]
    const int k = blockIdx.x;
    const int b = blockIdx.y;
    float m = -3.402823466e+38f;
    for (int t = threadIdx.x; t < TT; t += 256)
        m = fmaxf(m, frame_logits[((size_t)b * TT + t) * KK + k]);
#pragma unroll
    for (int off = 16; off > 0; off >>= 1)
        m = fmaxf(m, __shfl_xor(m, off, 32));
    __shared__ float red[8];
    if ((threadIdx.x & 31) == 0) red[threadIdx.x >> 5] = m;
    __syncthreads();
    if (threadIdx.x == 0) {
        float mm = red[0];
#pragma unroll
        for (int w = 1; w < 8; ++w) mm = fmaxf(mm, red[w]);
        clip_logits[(size_t)b * KK + k] = mm;
        clip_probs[(size_t)b * KK + k]  = sigmoid_f(mm);
    }
}

// ---------------------------------------------------------------------------
// Launch
// ---------------------------------------------------------------------------
extern "C" void kernel_launch(void* const* d_in, const int* in_sizes, int n_in,
                              void* d_out, int out_size, void* d_ws, size_t ws_size,
                              hipStream_t stream) {
    const float* fe     = (const float*)d_in[0];
    const float* conv_w = (const float*)d_in[1];
    const float* conv_b = (const float*)d_in[2];
    const float* W1     = (const float*)d_in[3];
    const float* b1     = (const float*)d_in[4];
    const float* W2     = (const float*)d_in[5];
    const float* b2     = (const float*)d_in[6];

    char* ws = (char*)d_ws;
    _Float16* feh    = (_Float16*)(ws);
    _Float16* Xt     = (_Float16*)(ws + FEH_BYTES);
    _Float16* wpack  = (_Float16*)(ws + FEH_BYTES + XT_BYTES);
    _Float16* w1pack = (_Float16*)(ws + FEH_BYTES + XT_BYTES + WPACK_BYTES);

    float* out          = (float*)d_out;
    float* clip_logits  = out;
    float* clip_probs   = out + (size_t)BB * KK;
    float* frame_logits = out + (size_t)2 * BB * KK;
    float* frame_probs  = frame_logits + (size_t)BB * TT * KK;

    // 0) cast + halo-pad embeddings
    {
        size_t total = (size_t)BB * TP * DD;
        int blocks = (int)((total + 255) / 256);
        cast_fe_kernel<<<blocks, 256, 0, stream>>>(fe, feh);
    }
    // 1) pack conv weights into A-fragment order ([j][di][hi])
    {
        size_t total = (size_t)3 * (HH/16) * (DD/32) * 512;
        int blocks = (int)((total + 255) / 256);
        pack_convw_kernel<<<blocks, 256, 0, stream>>>(conv_w, wpack);
    }
    // 2) pack W1 into A-fragment order
    {
        size_t total = (size_t)(KK*H2/16) * (HH/32) * 512;
        int blocks = (int)((total + 255) / 256);
        pack_w1_kernel<<<blocks, 256, 0, stream>>>(W1, w1pack);
    }
    // 3) conv + SiLU (WMMA, LDS-staged B, loads-first batches)
    conv_silu_kernel<<<dim3(NT_TILES, BB), 256, 0, stream>>>(feh, wpack, conv_b, Xt);
    // 4) K heads (WMMA) + fused SiLU + W2 + sigmoid
    heads_kernel<<<dim3(NT_TILES, BB), 256, 0, stream>>>(
        Xt, w1pack, b1, W2, b2, frame_logits, frame_probs);
    // 5) clip max + sigmoid
    clip_kernel<<<dim3(KK, BB), 256, 0, stream>>>(frame_logits, clip_logits, clip_probs);
}
